// DAGRNN_31774168056109
// MI455X (gfx1250) — matrix operand order repaired and checked by reading, not compiled, not run on hardware
//
#include <hip/hip_runtime.h>
#include <math.h>

// ---------------------------------------------------------------------------
// Types for CDNA5 WMMA (wave32): 16x16x32 bf16 -> f32
// ---------------------------------------------------------------------------
typedef __attribute__((ext_vector_type(16))) __bf16 v16bf;
typedef __attribute__((ext_vector_type(8)))  __bf16 v8bf;
typedef __attribute__((ext_vector_type(8)))  float  v8f;
typedef __attribute__((ext_vector_type(4)))  int    v4i;

union Frag { v16bf v; struct { v8bf lo; v8bf hi; } h; };

// Optional CDNA5 async global->LDS copy path (ASYNCcnt-tracked, ISA ch.10/15.18)
// Probe-confirmed signature: (v4i addrspace(1)*, v4i addrspace(3)*, imm, imm)
#if defined(__has_builtin)
#if __has_builtin(__builtin_amdgcn_global_load_async_to_lds_b128) && \
    __has_builtin(__builtin_amdgcn_s_wait_asynccnt)
#define USE_ASYNC_LDS 1
#endif
#endif

#if USE_ASYNC_LDS
typedef __attribute__((address_space(1))) v4i* g_v4i_ptr;   // global v4i*
typedef __attribute__((address_space(3))) v4i* l_v4i_ptr;   // LDS v4i*
#endif

// A-matrix 16x32 bf16, contiguous k order (B uses the same; not the ISA table order):
// lanes 0-15:  M=lane,   elems 0-15 -> K=k0+0..15
// lanes 16-31: M=lane-16,elems 0-15 -> K=k0+16..31
__device__ __forceinline__ v16bf load_a_frag(const __bf16* X, size_t lda,
                                             int m0, int k0, int lane) {
  const int half = lane >> 4;
  const int r    = lane & 15;
  const __bf16* p = X + (size_t)(m0 + r) * lda + (size_t)(k0 + (half ? 16 : 0));
  Frag f;
  f.h.lo = *reinterpret_cast<const v8bf*>(p);
  f.h.hi = *reinterpret_cast<const v8bf*>(p + 8);
  return f.v;
}

// B-matrix 32x16 bf16 (K x N), B element (k,n) = W[n,k], W row-major [N,K]:
// lanes 0-15:  N=lane,    K=k0+0..15 ; lanes 16-31: N=lane-16, K=k0+16..31
__device__ __forceinline__ v16bf load_b_frag(const __bf16* W, size_t ldw,
                                             int n0, int k0, int lane) {
  const int half = lane >> 4;
  const int c    = lane & 15;
  const __bf16* p = W + (size_t)(n0 + c) * ldw + (size_t)(k0 + (half ? 16 : 0));
  Frag f;
  f.h.lo = *reinterpret_cast<const v8bf*>(p);
  f.h.hi = *reinterpret_cast<const v8bf*>(p + 8);
  return f.v;
}

__device__ __forceinline__ v8f wmma_bf16(v16bf a, v16bf b, v8f c) {
  return __builtin_amdgcn_wmma_f32_16x16x32_bf16(
      /*neg_a=*/false, a, /*neg_b=*/false, b,
      /*c_mod=*/(short)0, c, /*reuse_a=*/false, /*reuse_b=*/false);
}

__device__ __forceinline__ float sigmoid_f(float x) {
  return 1.f / (1.f + __expf(-x));
}
// overflow-safe tanh
__device__ __forceinline__ float tanh_f(float x) {
  float t = __expf(-2.f * fabsf(x));
  float r = (1.f - t) / (1.f + t);
  return copysignf(r, x);
}

// ---------------------------------------------------------------------------
// f32 -> bf16 conversion
// ---------------------------------------------------------------------------
__global__ void __launch_bounds__(256)
f32_to_bf16_kernel(const float* __restrict__ src, __bf16* __restrict__ dst,
                   size_t n) {
  size_t i = (size_t)blockIdx.x * blockDim.x + threadIdx.x;
  if (i < n) dst[i] = (__bf16)src[i];
}

// ---------------------------------------------------------------------------
// Generic WMMA GEMM: Y[M,N] = act( X1 @ W1^T (+ X2 @ W2^T) + b1 (+ b2) )
// block = 128 threads = 4 waves; block tile = 128 rows x 16 cols
// ---------------------------------------------------------------------------
__global__ void __launch_bounds__(128)
wmma_gemm_kernel(const __bf16* __restrict__ X1, int lda1,
                 const __bf16* __restrict__ W1, int K1,
                 const __bf16* __restrict__ X2, int lda2,
                 const __bf16* __restrict__ W2, int K2,
                 const float* __restrict__ bias1,
                 const float* __restrict__ bias2,
                 int relu,
                 float* __restrict__ outF,
                 __bf16* __restrict__ outB, int ldo) {
  const int lane = threadIdx.x & 31;
  const int wave = threadIdx.x >> 5;
  const int m0 = blockIdx.x * 128 + wave * 32;
  const int n0 = blockIdx.y * 16;

  v8f acc0 = {};
  v8f acc1 = {};

  for (int k0 = 0; k0 < K1; k0 += 32) {
    v16bf b  = load_b_frag(W1, (size_t)K1, n0, k0, lane);
    v16bf a0 = load_a_frag(X1, (size_t)lda1, m0,      k0, lane);
    v16bf a1 = load_a_frag(X1, (size_t)lda1, m0 + 16, k0, lane);
    acc0 = wmma_bf16(a0, b, acc0);
    acc1 = wmma_bf16(a1, b, acc1);
  }
  if (X2) {
    for (int k0 = 0; k0 < K2; k0 += 32) {
      v16bf b  = load_b_frag(W2, (size_t)K2, n0, k0, lane);
      v16bf a0 = load_a_frag(X2, (size_t)lda2, m0,      k0, lane);
      v16bf a1 = load_a_frag(X2, (size_t)lda2, m0 + 16, k0, lane);
      acc0 = wmma_bf16(a0, b, acc0);
      acc1 = wmma_bf16(a1, b, acc1);
    }
  }

  const int col = n0 + (lane & 15);
  float bias = bias1[col];
  if (bias2) bias += bias2[col];

  // C/D layout: vgpr j -> M = j (lanes 0-15) / j+8 (lanes 16-31); N = lane&15
  for (int sub = 0; sub < 2; ++sub) {
    v8f acc = sub ? acc1 : acc0;
    int rbase = m0 + sub * 16 + ((lane < 16) ? 0 : 8);
    for (int j = 0; j < 8; ++j) {
      float v = acc[j] + bias;
      if (relu) v = fmaxf(v, 0.f);
      size_t idx = (size_t)(rbase + j) * (size_t)ldo + (size_t)col;
      if (outF) outF[idx] = v;
      if (outB) outB[idx] = (__bf16)v;
    }
  }
}

// ---------------------------------------------------------------------------
// Per-node prep: h_in/c_in = adjacency-weighted sum of stored states.
// Null node (row of zeros, i.e. weights sum to 0) -> copy (h0, c0).
// ---------------------------------------------------------------------------
__global__ void __launch_bounds__(256)
prep_kernel(const float* __restrict__ adj_row, int S,
            const float* __restrict__ h0f, const __bf16* __restrict__ h0b,
            const float* __restrict__ Hs, const float* __restrict__ Cs,
            __bf16* __restrict__ hinB, float* __restrict__ cinF, int BH) {
  int i = blockIdx.x * blockDim.x + threadIdx.x;
  if (i >= BH) return;
  float hs = 0.f, cs = 0.f, asum = 0.f;
  for (int j = 0; j < S; ++j) {
    float a = adj_row[j];
    if (a != 0.f) {
      asum += a;
      hs += a * Hs[(size_t)j * (size_t)BH + (size_t)i];
      cs += a * Cs[(size_t)j * (size_t)BH + (size_t)i];
    }
  }
  if (asum == 0.f) {  // null node: use (h0, c0)
    hinB[i] = h0b[i];
    cinF[i] = h0f[i];
  } else {
    hinB[i] = (__bf16)hs;
    cinF[i] = cs;
  }
}

// ---------------------------------------------------------------------------
// Fused LSTM node: g = x@W_ih^T + h_in@W_hh^T + b_ih + b_hh, then pointwise.
// Each wave: 32 rows x 16 hidden cols, computing all 4 gates for its cols
// (8 accumulators). Gates never leave registers.
//
// The block's weight tile (4 gates x 16 cols x 256 K = 32 KB bf16) is staged
// in LDS (async global->LDS when available), since it is identical for all 4
// waves of the block. Row pad of +8 bf16 gives a 132-dword row stride
// (== 4 mod 64 banks) -> conflict-free b128 DS reads per 16-lane half.
// ---------------------------------------------------------------------------
#define WT_LDK 264  // 256 + 8 pad, in bf16 elements

__global__ void __launch_bounds__(128)
lstm_step_kernel(const __bf16* __restrict__ Xs, int ldx,   // pipelines step slice
                 const __bf16* __restrict__ Hin,           // [B,H] bf16
                 const float* __restrict__ Cin,            // [B,H] f32
                 const __bf16* __restrict__ Wih,           // [4H,I] bf16
                 const __bf16* __restrict__ Whh,           // [4H,H] bf16
                 const float* __restrict__ bih,
                 const float* __restrict__ bhh,
                 float* __restrict__ Hout,                 // [B,H] f32 (state slot)
                 float* __restrict__ Cout) {
  const int H = 256, I = 256;
  const int lane = threadIdx.x & 31;
  const int wave = threadIdx.x >> 5;
  const int m0 = blockIdx.x * 128 + wave * 32;
  const int n0 = blockIdx.y * 16;

  __shared__ __bf16 Wt[64 * WT_LDK];  // 64 rows (4 gates x 16 cols) x 256 K

  // cooperative stage of one weight matrix tile into LDS:
  // LDS row r (0..63): gate g=r>>4, col c=r&15 -> global row g*H + n0 + c.
  // 128 threads: thread t covers row t>>1, K-half (t&1)*128, 16 b128 chunks.
  auto stage_tile = [&](const __bf16* __restrict__ W) {
    const int t = threadIdx.x;
    const int r = t >> 1;
    const int kh = (t & 1) * 128;
    const int grow = (r >> 4) * H + n0 + (r & 15);
    const __bf16* src = W + (size_t)grow * 256 + kh;
    __bf16* dst = &Wt[(size_t)r * WT_LDK + kh];
#if USE_ASYNC_LDS
    #pragma unroll
    for (int i = 0; i < 16; ++i) {
      __builtin_amdgcn_global_load_async_to_lds_b128(
          (g_v4i_ptr)(src + i * 8), (l_v4i_ptr)(dst + i * 8), 0, 0);
    }
    __builtin_amdgcn_s_wait_asynccnt(0);
#else
    #pragma unroll
    for (int i = 0; i < 16; ++i)
      *reinterpret_cast<v8bf*>(dst + i * 8) =
          *reinterpret_cast<const v8bf*>(src + i * 8);
#endif
  };

  // B fragment from the staged LDS tile (K x 16 for gate g at local k0)
  auto lds_b_frag = [&](int g, int k0) -> v16bf {
    const int half = lane >> 4;
    const int c    = lane & 15;
    const __bf16* p = &Wt[(size_t)(g * 16 + c) * WT_LDK + k0 + (half ? 16 : 0)];
    Frag f;
    f.h.lo = *reinterpret_cast<const v8bf*>(p);
    f.h.hi = *reinterpret_cast<const v8bf*>(p + 8);
    return f.v;
  };

  v8f acc[4][2];
  for (int g = 0; g < 4; ++g) { acc[g][0] = (v8f){}; acc[g][1] = (v8f){}; }

  // phase 1: x @ W_ih^T
  stage_tile(Wih);
  __syncthreads();
  for (int k0 = 0; k0 < I; k0 += 32) {
    v16bf a0 = load_a_frag(Xs, (size_t)ldx, m0,      k0, lane);
    v16bf a1 = load_a_frag(Xs, (size_t)ldx, m0 + 16, k0, lane);
    for (int g = 0; g < 4; ++g) {
      v16bf b = lds_b_frag(g, k0);
      acc[g][0] = wmma_bf16(a0, b, acc[g][0]);
      acc[g][1] = wmma_bf16(a1, b, acc[g][1]);
    }
  }

  // phase 2: h_in @ W_hh^T (re-stage LDS with the recurrent weights)
  __syncthreads();
  stage_tile(Whh);
  __syncthreads();
  for (int k0 = 0; k0 < H; k0 += 32) {
    v16bf a0 = load_a_frag(Hin, (size_t)H, m0,      k0, lane);
    v16bf a1 = load_a_frag(Hin, (size_t)H, m0 + 16, k0, lane);
    for (int g = 0; g < 4; ++g) {
      v16bf b = lds_b_frag(g, k0);
      acc[g][0] = wmma_bf16(a0, b, acc[g][0]);
      acc[g][1] = wmma_bf16(a1, b, acc[g][1]);
    }
  }

  const int col = n0 + (lane & 15);
  float bsum[4];
  for (int g = 0; g < 4; ++g) bsum[g] = bih[g * H + col] + bhh[g * H + col];

  for (int sub = 0; sub < 2; ++sub) {
    int rbase = m0 + sub * 16 + ((lane < 16) ? 0 : 8);
    for (int j = 0; j < 8; ++j) {
      int row = rbase + j;
      size_t idx = (size_t)row * (size_t)H + (size_t)col;
      float gi = sigmoid_f(acc[0][sub][j] + bsum[0]);
      float gf = sigmoid_f(acc[1][sub][j] + bsum[1]);
      float gg = tanh_f  (acc[2][sub][j] + bsum[2]);
      float go = sigmoid_f(acc[3][sub][j] + bsum[3]);
      float cn = gf * Cin[idx] + gi * gg;
      float hn = go * tanh_f(cn);
      Cout[idx] = cn;
      Hout[idx] = hn;
    }
  }
}

// ---------------------------------------------------------------------------
// Final N=1 projection: out[b] = y1[b,:]@out_w2 + b2 + h_last[b,:]@out_skip_w + sb
// One wave32 per output row, shuffle reduction.
// ---------------------------------------------------------------------------
__global__ void __launch_bounds__(256)
final_kernel(const float* __restrict__ y1, const float* __restrict__ hlast,
             const float* __restrict__ w2, const float* __restrict__ b2,
             const float* __restrict__ sw, const float* __restrict__ sb,
             float* __restrict__ out, int B, int H) {
  int wavesPerBlock = blockDim.x >> 5;
  int row  = blockIdx.x * wavesPerBlock + (threadIdx.x >> 5);
  int lane = threadIdx.x & 31;
  if (row >= B) return;
  float s = 0.f;
  for (int k = lane; k < H; k += 32) {
    size_t idx = (size_t)row * (size_t)H + (size_t)k;
    s += y1[idx] * w2[k] + hlast[idx] * sw[k];
  }
  for (int off = 16; off > 0; off >>= 1) s += __shfl_xor(s, off, 32);
  if (lane == 0) out[row] = s + b2[0] + sb[0];
}

// ---------------------------------------------------------------------------
// Host-side orchestration (graph-capture safe: launches only)
// ---------------------------------------------------------------------------
extern "C" void kernel_launch(void* const* d_in, const int* in_sizes, int n_in,
                              void* d_out, int out_size, void* d_ws, size_t ws_size,
                              hipStream_t stream) {
  (void)in_sizes; (void)n_in; (void)out_size; (void)ws_size;
  const int B = 4096, S = 32, I = 256, H = 256, MF = 128;

  const float* pipelines  = (const float*)d_in[0];
  const float* metafeat   = (const float*)d_in[1];
  const float* adj        = (const float*)d_in[2];
  // d_in[3] is_null: not needed (null node == all-zero adjacency row)
  const float* in_w1      = (const float*)d_in[4];
  const float* in_b1      = (const float*)d_in[5];
  const float* in_w2      = (const float*)d_in[6];
  const float* in_b2      = (const float*)d_in[7];
  const float* in_skip_w  = (const float*)d_in[8];
  const float* in_skip_b  = (const float*)d_in[9];
  const float* W_ih       = (const float*)d_in[10];
  const float* W_hh       = (const float*)d_in[11];
  const float* b_ih       = (const float*)d_in[12];
  const float* b_hh       = (const float*)d_in[13];
  const float* out_w1     = (const float*)d_in[14];
  const float* out_b1     = (const float*)d_in[15];
  const float* out_w2     = (const float*)d_in[16];
  const float* out_b2     = (const float*)d_in[17];
  const float* out_skip_w = (const float*)d_in[18];
  const float* out_skip_b = (const float*)d_in[19];

  char* ws = (char*)d_ws;
  size_t off = 0;
  auto carve = [&](size_t bytes) -> char* {
    char* p = ws + off;
    off += (bytes + 255) & ~(size_t)255;
    return p;
  };

  const size_t BH = (size_t)B * H;
  __bf16* pipeB  = (__bf16*)carve((size_t)B * S * I * 2);   // bf16 pipelines
  float*  Hs     = (float*) carve((size_t)S * BH * 4);      // per-node h states
  float*  Cs     = (float*) carve((size_t)S * BH * 4);      // per-node c states
  float*  h0f    = (float*) carve(BH * 4);
  __bf16* h0b    = (__bf16*)carve(BH * 2);
  __bf16* hinB   = (__bf16*)carve(BH * 2);
  float*  cinF   = (float*) carve(BH * 4);
  __bf16* tB     = (__bf16*)carve(BH * 2);                  // input MLP hidden
  __bf16* mfB    = (__bf16*)carve((size_t)B * MF * 2);
  __bf16* hlastB = (__bf16*)carve(BH * 2);
  float*  y1f    = (float*) carve(BH * 4);
  __bf16* WihB   = (__bf16*)carve((size_t)4 * H * I * 2);
  __bf16* WhhB   = (__bf16*)carve((size_t)4 * H * H * 2);
  __bf16* w1B    = (__bf16*)carve((size_t)H * MF * 2);
  __bf16* w2B    = (__bf16*)carve((size_t)H * H * 2);
  __bf16* swB    = (__bf16*)carve((size_t)H * MF * 2);
  __bf16* ow1B   = (__bf16*)carve((size_t)H * H * 2);

  auto conv = [&](const float* src, __bf16* dst, size_t n) {
    int blocks = (int)((n + 255) / 256);
    f32_to_bf16_kernel<<<blocks, 256, 0, stream>>>(src, dst, n);
  };

  // Precision conversions (weights stay L2-resident afterwards)
  conv(pipelines, pipeB, (size_t)B * S * I);
  conv(metafeat,  mfB,   (size_t)B * MF);
  conv(W_ih,      WihB,  (size_t)4 * H * I);
  conv(W_hh,      WhhB,  (size_t)4 * H * H);
  conv(in_w1,     w1B,   (size_t)H * MF);
  conv(in_w2,     w2B,   (size_t)H * H);
  conv(in_skip_w, swB,   (size_t)H * MF);
  conv(out_w1,    ow1B,  (size_t)H * H);

  dim3 blk(128);
  dim3 grid(B / 128, H / 16);

  // Input MLP: t = relu(mf @ w1^T + b1)
  wmma_gemm_kernel<<<grid, blk, 0, stream>>>(
      mfB, MF, w1B, MF, nullptr, 0, nullptr, 0,
      in_b1, nullptr, /*relu=*/1, nullptr, tB, H);
  // h0 = c0 = relu(t @ w2^T + b2 + mf @ skip^T + sb)
  wmma_gemm_kernel<<<grid, blk, 0, stream>>>(
      tB, H, w2B, H, mfB, MF, swB, MF,
      in_b2, in_skip_b, /*relu=*/1, h0f, h0b, H);

  // Sequential DAG-LSTM over S nodes
  int prepBlocks = (int)((BH + 255) / 256);
  for (int s = 0; s < S; ++s) {
    prep_kernel<<<prepBlocks, 256, 0, stream>>>(
        adj + (size_t)s * S, S, h0f, h0b, Hs, Cs, hinB, cinF, (int)BH);
    lstm_step_kernel<<<grid, blk, 0, stream>>>(
        pipeB + (size_t)s * I, S * I, hinB, cinF,
        WihB, WhhB, b_ih, b_hh,
        Hs + (size_t)s * BH, Cs + (size_t)s * BH);
  }

  // Output MLP stage 1: y1 = relu(h_last @ out_w1^T + out_b1)
  const float* hlastF = Hs + (size_t)(S - 1) * BH;
  conv(hlastF, hlastB, BH);
  wmma_gemm_kernel<<<grid, blk, 0, stream>>>(
      hlastB, H, ow1B, H, nullptr, 0, nullptr, 0,
      out_b1, nullptr, /*relu=*/1, y1f, nullptr, H);

  // Output MLP stage 2 (N=1 dot products, f32)
  final_kernel<<<(B + 7) / 8, 256, 0, stream>>>(
      y1f, hlastF, out_w2, out_b2, out_skip_w, out_skip_b,
      (float*)d_out, B, H);
}
